// MultiHeadLatentAttentionDecodeLayer_16965120819285
// MI455X (gfx1250) — compile-verified, matrix operand order checked by reading, not run on hardware
//
#include <hip/hip_runtime.h>
#include <math.h>

// ---------------------------------------------------------------------------
// MLA decode for MI455X (gfx1250, wave32, WMMA).
//
// Roofline: ~150 MB mandatory HBM reads (~6.4us @ 23.3 TB/s) vs 18.3 GFLOP.
// fp32 WMMA (16x16x4) would be compute-bound; converting streamed data to
// bf16 in LDS and using v_wmma_f32_16x16x32_bf16 (f32 accumulate) makes the
// kernel memory-bound => optimal for this chip.
//
// Kernel 1: flash-decode partial attention, split-K over the 4096 keys.
//   grid = (NSPLIT, H/64, B), block = 512 threads = 16 waves.
//   Per CTA: 64 heads x 512 keys, streamed in 64-key LDS tiles.
//   LDS holds bf16 Q [64x584], K row-major [64x584] (QK A-frags),
//   K transposed [576x72] (PV B-frags), P [64x72], softmax stats.
// Kernel 2: log-sum-exp combine of the NSPLIT partials.
// ---------------------------------------------------------------------------

#define BDIM   16
#define HDIM   128
#define SDIM   4096
#define DNOPE  512
#define DPE    64
#define DTOT   576                  // 512 + 64
#define NSPLIT 8
#define CHUNK  (SDIM / NSPLIT)      // 512 keys per CTA
#define KEY_TILE 64
#define NTILES (CHUNK / KEY_TILE)   // 8
#define HEADS_CTA 64
#define KSTEPS (DTOT / 32)          // 18 wmma k-steps
#define QS_STRIDE 584               // 576 + 8 pad (1168B rows, 16B aligned)
#define KST_STRIDE 72               // 64 + 8 pad  (144B rows, 16B aligned)

typedef __attribute__((ext_vector_type(16))) __bf16 bf16x16;
typedef __attribute__((ext_vector_type(8)))  float  f32x8;

union FragU {
  uint4   q[2];
  bf16x16 v;
};

__device__ __forceinline__ unsigned short f2bf(float x) {
  unsigned u = __float_as_uint(x);
  u += 0x7FFFu + ((u >> 16) & 1u);        // round-to-nearest-even
  return (unsigned short)(u >> 16);
}

__global__ __launch_bounds__(512) void mla_partial_kernel(
    const float* __restrict__ q, const float* __restrict__ qpe,
    const float* __restrict__ k, const float* __restrict__ kpe,
    float* __restrict__ Opart, float* __restrict__ mPart,
    float* __restrict__ lPart) {
  const int j   = blockIdx.x;             // key split
  const int hh  = blockIdx.y;             // head half (0/1)
  const int b   = blockIdx.z;             // batch
  const int tid = threadIdx.x;
  const int wave = tid >> 5;
  const int lane = tid & 31;
  const int hg = wave >> 2;               // head group 0..3 (16 heads each)
  const int kg = wave & 3;                // key subtile 0..3 (QK) / dim group (PV)
  const int lrow = lane & 15;
  const int halfsel = (lane >> 4) << 3;   // 0 or 8 (16-bit frag lane-half K offset)

  __shared__ __align__(16) unsigned short Qs[HEADS_CTA][QS_STRIDE];   // bf16
  __shared__ __align__(16) unsigned short Ks[KEY_TILE][QS_STRIDE];    // bf16 [key][dim]
  __shared__ __align__(16) unsigned short Kst[DTOT][KST_STRIDE];      // bf16 [dim][key]
  __shared__ __align__(16) unsigned short Pt[HEADS_CTA][KST_STRIDE];  // bf16 [head][key]
  __shared__ float mS[HEADS_CTA], lS[HEADS_CTA], cS[HEADS_CTA];
  __shared__ float redmax[4][16][4];      // [hg][head-in-group][kg]

  const float scale = 0.041666666666666664f;   // 1/sqrt(576)

  // ---- Phase A: load Q (pre-scaled) -> LDS bf16 -------------------------
  {
    const float* qb = q + ((long long)b * HDIM + hh * HEADS_CTA) * DNOPE;
    for (int i = tid; i < HEADS_CTA * (DNOPE / 4); i += 512) {
      const int h = i >> 7, dv = (i & 127) << 2;
      float4 v = *(const float4*)(qb + (long long)h * DNOPE + dv);
      ushort4 s;
      s.x = f2bf(v.x * scale); s.y = f2bf(v.y * scale);
      s.z = f2bf(v.z * scale); s.w = f2bf(v.w * scale);
      *(ushort4*)&Qs[h][dv] = s;
    }
    const float* qpb = qpe + ((long long)b * HDIM + hh * HEADS_CTA) * DPE;
    for (int i = tid; i < HEADS_CTA * (DPE / 4); i += 512) {
      const int h = i >> 4, dv = (i & 15) << 2;
      float4 v = *(const float4*)(qpb + (long long)h * DPE + dv);
      ushort4 s;
      s.x = f2bf(v.x * scale); s.y = f2bf(v.y * scale);
      s.z = f2bf(v.z * scale); s.w = f2bf(v.w * scale);
      *(ushort4*)&Qs[h][DNOPE + dv] = s;
    }
  }
  if (tid < HEADS_CTA) { mS[tid] = -INFINITY; lS[tid] = 0.0f; }

  f32x8 O[8] = {};                        // 16 heads x 128 dims accumulator

  const float* kb  = k   + (long long)b * SDIM * DNOPE;
  const float* kpb = kpe + (long long)b * SDIM * DPE;

  for (int t = 0; t < NTILES; ++t) {
    const int sbase = j * CHUNK + t * KEY_TILE;
    __syncthreads();                       // protect Ks/Kst/Pt reuse

    // ---- load 64-key K tile -> bf16, both layouts ----------------------
    for (int i = tid; i < KEY_TILE * (DNOPE / 4); i += 512) {
      const int kk = i >> 7, dv = (i & 127) << 2;
      float4 v = *(const float4*)(kb + (long long)(sbase + kk) * DNOPE + dv);
      ushort4 s;
      s.x = f2bf(v.x); s.y = f2bf(v.y); s.z = f2bf(v.z); s.w = f2bf(v.w);
      *(ushort4*)&Ks[kk][dv] = s;
      Kst[dv + 0][kk] = s.x; Kst[dv + 1][kk] = s.y;
      Kst[dv + 2][kk] = s.z; Kst[dv + 3][kk] = s.w;
    }
    for (int i = tid; i < KEY_TILE * (DPE / 4); i += 512) {
      const int kk = i >> 4, dv = (i & 15) << 2;
      float4 v = *(const float4*)(kpb + (long long)(sbase + kk) * DPE + dv);
      ushort4 s;
      s.x = f2bf(v.x); s.y = f2bf(v.y); s.z = f2bf(v.z); s.w = f2bf(v.w);
      *(ushort4*)&Ks[kk][DNOPE + dv] = s;
      Kst[DNOPE + dv + 0][kk] = s.x; Kst[DNOPE + dv + 1][kk] = s.y;
      Kst[DNOPE + dv + 2][kk] = s.z; Kst[DNOPE + dv + 3][kk] = s.w;
    }
    // prefetch next key tile into L2 while computing this one
    if (t + 1 < NTILES) {
      __builtin_prefetch(kb + (long long)(sbase + KEY_TILE) * DNOPE + tid * 16, 0, 0);
    }
    __syncthreads();

    // ---- QK: S^T tile [16 keys x 16 heads] = K_tile x Q^T ---------------
    f32x8 acc = {};
    const int krow = kg * 16 + lrow;
    const int hrow = hg * 16 + lrow;
#pragma unroll
    for (int kk = 0; kk < KSTEPS; ++kk) {
      FragU a, bq;
      const int d0 = kk * 32 + halfsel;
      a.q[0]  = *(const uint4*)&Ks[krow][d0];
      a.q[1]  = *(const uint4*)&Ks[krow][d0 + 16];
      bq.q[0] = *(const uint4*)&Qs[hrow][d0];
      bq.q[1] = *(const uint4*)&Qs[hrow][d0 + 16];
      acc = __builtin_amdgcn_wmma_f32_16x16x32_bf16(
          false, a.v, false, bq.v, (short)0, acc, false, false);
    }

    // ---- online softmax: per-head tile max ------------------------------
    float tmax = acc[0];
#pragma unroll
    for (int r = 1; r < 8; ++r) tmax = fmaxf(tmax, acc[r]);
    tmax = fmaxf(tmax, __shfl_xor(tmax, 16, 32));
    if (lane < 16) redmax[hg][lane][kg] = tmax;
    __syncthreads();

    if (kg == 0 && lane < 16) {
      const int hl = hg * 16 + lane;
      float mt = fmaxf(fmaxf(redmax[hg][lane][0], redmax[hg][lane][1]),
                       fmaxf(redmax[hg][lane][2], redmax[hg][lane][3]));
      float mo = mS[hl];
      float mn = fmaxf(mo, mt);
      float c  = __expf(mo - mn);          // 0 for first tile (mo = -inf)
      mS[hl] = mn; cS[hl] = c; lS[hl] *= c;
    }
    __syncthreads();

    // ---- exponentiate, accumulate row sums, pack P -> bf16 LDS ----------
    {
      const float mval = mS[hrow];
      float p[8], sum = 0.0f;
#pragma unroll
      for (int r = 0; r < 8; ++r) { p[r] = __expf(acc[r] - mval); sum += p[r]; }
      sum += __shfl_xor(sum, 16, 32);
      if (lane < 16) atomicAdd(&lS[hrow], sum);   // ds_add_f32
      uint4 pk;
      pk.x = (unsigned)f2bf(p[0]) | ((unsigned)f2bf(p[1]) << 16);
      pk.y = (unsigned)f2bf(p[2]) | ((unsigned)f2bf(p[3]) << 16);
      pk.z = (unsigned)f2bf(p[4]) | ((unsigned)f2bf(p[5]) << 16);
      pk.w = (unsigned)f2bf(p[6]) | ((unsigned)f2bf(p[7]) << 16);
      *(uint4*)&Pt[hrow][kg * 16 + halfsel] = pk;
    }
    __syncthreads();

    // ---- PV: O[16h x 128d] = c*O + P[16h x 64k] x V[64k x 16d] ----------
    {
      float cv[8];
#pragma unroll
      for (int r = 0; r < 8; ++r) cv[r] = cS[hg * 16 + r + halfsel];
#pragma unroll
      for (int nt = 0; nt < 8; ++nt)
#pragma unroll
        for (int r = 0; r < 8; ++r) O[nt][r] *= cv[r];

      const int drow = kg * 128 + lrow;    // kg reused as dim group
#pragma unroll
      for (int nt = 0; nt < 8; ++nt) {
#pragma unroll
        for (int ks2 = 0; ks2 < 2; ++ks2) {
          FragU pa, vb;
          const int kb0 = ks2 * 32 + halfsel;
          pa.q[0] = *(const uint4*)&Pt[hrow][kb0];
          pa.q[1] = *(const uint4*)&Pt[hrow][kb0 + 16];
          vb.q[0] = *(const uint4*)&Kst[drow + nt * 16][kb0];
          vb.q[1] = *(const uint4*)&Kst[drow + nt * 16][kb0 + 16];
          O[nt] = __builtin_amdgcn_wmma_f32_16x16x32_bf16(
              false, pa.v, false, vb.v, (short)0, O[nt], false, false);
        }
      }
    }
  }
  __syncthreads();

  // ---- write split partials (unnormalized O, m, l) ----------------------
  {
    const long long base = ((long long)j * BDIM + b) * HDIM + hh * HEADS_CTA;
#pragma unroll
    for (int nt = 0; nt < 8; ++nt) {
#pragma unroll
      for (int r = 0; r < 8; ++r) {
        const int hl = hg * 16 + r + halfsel;
        const int d  = kg * 128 + nt * 16 + lrow;
        Opart[(base + hl) * DNOPE + d] = O[nt][r];
      }
    }
    if (tid < HEADS_CTA) {
      mPart[base + tid] = mS[tid];
      lPart[base + tid] = lS[tid];
    }
  }
}

// ---------------------------------------------------------------------------
// Kernel 2: combine NSPLIT partials with log-sum-exp weighting.
// ---------------------------------------------------------------------------
__global__ __launch_bounds__(128) void mla_combine_kernel(
    const float* __restrict__ Opart, const float* __restrict__ mPart,
    const float* __restrict__ lPart, float* __restrict__ out) {
  const int h = blockIdx.x, b = blockIdx.y;
  const int tid = threadIdx.x;

  float mj[NSPLIT], lj[NSPLIT];
  float m = -INFINITY;
#pragma unroll
  for (int jj = 0; jj < NSPLIT; ++jj) {
    const long long idx = ((long long)jj * BDIM + b) * HDIM + h;
    mj[jj] = mPart[idx];
    lj[jj] = lPart[idx];
    m = fmaxf(m, mj[jj]);
  }
  float wj[NSPLIT], W = 0.0f;
#pragma unroll
  for (int jj = 0; jj < NSPLIT; ++jj) {
    wj[jj] = __expf(mj[jj] - m);
    W += wj[jj] * lj[jj];
  }
  const float inv = 1.0f / W;

  const int d = tid * 4;                  // 128 threads x 4 = 512 dims
  float4 acc = {0.0f, 0.0f, 0.0f, 0.0f};
#pragma unroll
  for (int jj = 0; jj < NSPLIT; ++jj) {
    const long long idx =
        (((long long)jj * BDIM + b) * HDIM + h) * DNOPE + d;
    float4 v = *(const float4*)(Opart + idx);
    acc.x += wj[jj] * v.x; acc.y += wj[jj] * v.y;
    acc.z += wj[jj] * v.z; acc.w += wj[jj] * v.w;
  }
  float4 o;
  o.x = acc.x * inv; o.y = acc.y * inv; o.z = acc.z * inv; o.w = acc.w * inv;
  *(float4*)(out + ((long long)b * HDIM + h) * DNOPE + d) = o;
}

extern "C" void kernel_launch(void* const* d_in, const int* in_sizes, int n_in,
                              void* d_out, int out_size, void* d_ws,
                              size_t ws_size, hipStream_t stream) {
  const float* q   = (const float*)d_in[0];
  const float* qpe = (const float*)d_in[1];
  const float* k   = (const float*)d_in[2];
  const float* kpe = (const float*)d_in[3];

  float* Opart = (float*)d_ws;  // [NSPLIT][B][H][512]
  float* mPart = Opart + (long long)NSPLIT * BDIM * HDIM * DNOPE;
  float* lPart = mPart + (long long)NSPLIT * BDIM * HDIM;

  dim3 g1(NSPLIT, HDIM / HEADS_CTA, BDIM);
  mla_partial_kernel<<<g1, 512, 0, stream>>>(q, qpe, k, kpe, Opart, mPart,
                                             lPart);
  dim3 g2(HDIM, BDIM);
  mla_combine_kernel<<<g2, 128, 0, stream>>>(Opart, mPart, lPart,
                                             (float*)d_out);
}
